// AggregateOrExclusive_16535624090065
// MI455X (gfx1250) — matrix operand order repaired and checked by reading, not compiled, not run on hardware
//
#include <hip/hip_runtime.h>

// ---------------- problem constants ----------------
#define E_EXP   8
#define N_TOK   8192
#define D_IN    1024
#define H_MID   2048
#define O_OUT   1024
#define OUTW    (O_OUT + E_EXP)           // 1032
#define MAX_TILES (N_TOK/32 + E_EXP)      // 264 worst-case 32-row tiles

typedef __attribute__((ext_vector_type(16))) __bf16 v16bf;
typedef __attribute__((ext_vector_type(8)))  float  v8f;

struct BPair { uint4 lo, hi; };           // 32B, same size as v16bf
struct F8    { float f[8]; };             // 32B, same size as v8f

__device__ __forceinline__ unsigned short bf16_rne(float f) {
  union { float f; unsigned int u; } v; v.f = f;
  unsigned int r = 0x7FFFu + ((v.u >> 16) & 1u);
  v.u += r;
  return (unsigned short)(v.u >> 16);
}

// ---------------- routing ----------------
__global__ void k_init_cnt(int* cnt) {
  if (threadIdx.x < E_EXP) cnt[threadIdx.x] = 0;
}

__global__ void k_count(const int* __restrict__ eidx, int* cnt, int* slot) {
  int n = blockIdx.x * blockDim.x + threadIdx.x;
  if (n < N_TOK) slot[n] = atomicAdd(&cnt[eidx[n]], 1);
}

__global__ void k_scan(const int* __restrict__ cnt, int* off, int* tOff) {
  if (threadIdx.x == 0) {
    int a = 0, t = 0;
    for (int e = 0; e < E_EXP; ++e) {
      off[e] = a; tOff[e] = t;
      int c = cnt[e];
      a += c; t += (c + 31) >> 5;
    }
    off[E_EXP] = a; tOff[E_EXP] = t;
  }
}

__global__ void k_scatter(const int* __restrict__ eidx, const int* __restrict__ slot,
                          const int* __restrict__ off, int* perm) {
  int n = blockIdx.x * blockDim.x + threadIdx.x;
  if (n < N_TOK) perm[off[eidx[n]] + slot[n]] = n;
}

// ---------------- weight convert + transpose (fp32 -> bf16) ----------------
// W1 [E,D,H] -> W1t [E,H,D]
__global__ void k_cvt_w1(const float* __restrict__ W1, unsigned short* __restrict__ W1t) {
  int i = blockIdx.x * blockDim.x + threadIdx.x;      // over E*H*D outputs
  int d = i & (D_IN - 1);
  int r = i / D_IN;
  int h = r & (H_MID - 1);
  int e = r / H_MID;
  W1t[i] = bf16_rne(W1[((size_t)e * D_IN + d) * H_MID + h]);
}

// W2 [E,H,O] -> W2t [E,O,H]
__global__ void k_cvt_w2(const float* __restrict__ W2, unsigned short* __restrict__ W2t) {
  int i = blockIdx.x * blockDim.x + threadIdx.x;      // over E*O*H outputs
  int h = i & (H_MID - 1);
  int r = i / H_MID;
  int o = r & (O_OUT - 1);
  int e = r / O_OUT;
  W2t[i] = bf16_rne(W2[((size_t)e * H_MID + h) * O_OUT + o]);
}

// ---------------- GEMM1: hid[pos,:] = bf16( x[tok] @ W1[e] + b1[e] ) ----------------
// block = 256 thr = 8 wave32; block tile = 32 rows x 512 cols; wave tile = 32x64.
__global__ __launch_bounds__(256) void k_gemm1(
    const float* __restrict__ x, const unsigned short* __restrict__ W1t,
    const float* __restrict__ b1, const int* __restrict__ perm,
    const int* __restrict__ off, const int* __restrict__ tOff,
    unsigned short* __restrict__ hid)
{
  __shared__ int sMeta[4];
  __shared__ int sTok[32];
  __shared__ __align__(16) unsigned short sA[32 * 32];   // 2KB bf16 tile

  const int tid  = threadIdx.x;
  const int lane = tid & 31;
  const int wave = tid >> 5;
  const int t    = blockIdx.y;

  if (tid == 0) {
    if (t >= tOff[E_EXP]) { sMeta[0] = -1; }
    else {
      int e = 0;
      while (!(t >= tOff[e] && t < tOff[e + 1])) ++e;
      int lt   = t - tOff[e];
      int base = off[e];
      int ne   = off[e + 1] - base;
      sMeta[0] = e;
      sMeta[1] = base + lt * 32;                 // first compacted row of tile
      sMeta[2] = (ne - lt * 32 < 32) ? (ne - lt * 32) : 32;
    }
  }
  __syncthreads();
  if (sMeta[0] < 0) return;
  const int e = sMeta[0], rowBase = sMeta[1], nRows = sMeta[2];
  if (tid < 32) sTok[tid] = perm[rowBase + ((tid < nRows) ? tid : 0)];

  const int h8 = lane >> 4;
  const int cn = lane & 15;
  const int colBase = blockIdx.x * 512 + wave * 64;

  v8f accL[4], accH[4];
#pragma unroll
  for (int s = 0; s < 4; ++s) {
    float b = b1[e * H_MID + colBase + s * 16 + cn];
    accL[s] = (v8f){b, b, b, b, b, b, b, b};
    accH[s] = accL[s];
  }
  __syncthreads();

  for (int kb = 0; kb < D_IN; kb += 32) {
    {   // stage 32x32 A tile into LDS as bf16 (gathered token rows), float4 loads
      int i0 = tid * 4;
      int r  = i0 >> 5, k = i0 & 31;
      float4 xv = *(const float4*)&x[(size_t)sTok[r] * D_IN + kb + k];
      unsigned int p0 = (unsigned)bf16_rne(xv.x) | ((unsigned)bf16_rne(xv.y) << 16);
      unsigned int p1 = (unsigned)bf16_rne(xv.z) | ((unsigned)bf16_rne(xv.w) << 16);
      uint2 pk; pk.x = p0; pk.y = p1;
      *(uint2*)&sA[r * 32 + k] = pk;
    }
    __syncthreads();

    BPair ap, ahp;
    ap.lo  = *(const uint4*)&sA[cn * 32 + h8 * 8];
    ap.hi  = *(const uint4*)&sA[cn * 32 + 16 + h8 * 8];
    ahp.lo = *(const uint4*)&sA[(16 + cn) * 32 + h8 * 8];
    ahp.hi = *(const uint4*)&sA[(16 + cn) * 32 + 16 + h8 * 8];
    v16bf aLo = __builtin_bit_cast(v16bf, ap);
    v16bf aHi = __builtin_bit_cast(v16bf, ahp);

    // preload all 4 B fragments, then burst 8 WMMAs (B reused for both row groups)
    v16bf bF[4];
#pragma unroll
    for (int s = 0; s < 4; ++s) {
      const unsigned short* bp =
          W1t + ((size_t)(e * H_MID + colBase + s * 16 + cn)) * D_IN + kb + h8 * 8;
      __builtin_prefetch((const void*)(bp + 32), 0, 0);
      BPair bb;
      bb.lo = *(const uint4*)bp;
      bb.hi = *(const uint4*)(bp + 16);
      bF[s] = __builtin_bit_cast(v16bf, bb);
    }
#pragma unroll
    for (int s = 0; s < 4; ++s) {
      accL[s] = __builtin_amdgcn_wmma_f32_16x16x32_bf16(
          false, aLo, false, bF[s], (short)0, accL[s], false, false);
      accH[s] = __builtin_amdgcn_wmma_f32_16x16x32_bf16(
          false, aHi, false, bF[s], (short)0, accH[s], false, false);
    }
    __syncthreads();
  }

#pragma unroll
  for (int s = 0; s < 4; ++s) {
    F8 frL = __builtin_bit_cast(F8, accL[s]);
    F8 frH = __builtin_bit_cast(F8, accH[s]);
    int hcol = colBase + s * 16 + cn;
#pragma unroll
    for (int i = 0; i < 8; ++i) {
      int rL = i + 8 * h8;
      int rH = 16 + rL;
      if (rL < nRows)
        hid[(size_t)(rowBase + rL) * H_MID + hcol] = bf16_rne(frL.f[i]);
      if (rH < nRows)
        hid[(size_t)(rowBase + rH) * H_MID + hcol] = bf16_rne(frH.f[i]);
    }
  }
}

// ---------------- GEMM2: out[tok, :O] = hid[pos] @ W2[e] + b2[e] ----------------
__global__ __launch_bounds__(256) void k_gemm2(
    const unsigned short* __restrict__ hid, const unsigned short* __restrict__ W2t,
    const float* __restrict__ b2, const int* __restrict__ perm,
    const int* __restrict__ off, const int* __restrict__ tOff,
    float* __restrict__ out)
{
  __shared__ int sMeta[4];
  __shared__ int sTok[32];
  __shared__ __align__(16) unsigned short sA[32 * 32];

  const int tid  = threadIdx.x;
  const int lane = tid & 31;
  const int wave = tid >> 5;
  const int t    = blockIdx.y;

  if (tid == 0) {
    if (t >= tOff[E_EXP]) { sMeta[0] = -1; }
    else {
      int e = 0;
      while (!(t >= tOff[e] && t < tOff[e + 1])) ++e;
      int lt   = t - tOff[e];
      int base = off[e];
      int ne   = off[e + 1] - base;
      sMeta[0] = e;
      sMeta[1] = base + lt * 32;
      sMeta[2] = (ne - lt * 32 < 32) ? (ne - lt * 32) : 32;
    }
  }
  __syncthreads();
  if (sMeta[0] < 0) return;
  const int e = sMeta[0], rowBase = sMeta[1], nRows = sMeta[2];
  if (tid < 32) sTok[tid] = perm[rowBase + ((tid < nRows) ? tid : 0)];

  const int h8 = lane >> 4;
  const int cn = lane & 15;
  const int colBase = blockIdx.x * 512 + wave * 64;

  v8f accL[4], accH[4];
#pragma unroll
  for (int s = 0; s < 4; ++s) {
    float b = b2[e * O_OUT + colBase + s * 16 + cn];
    accL[s] = (v8f){b, b, b, b, b, b, b, b};
    accH[s] = accL[s];
  }
  __syncthreads();

  for (int kb = 0; kb < H_MID; kb += 32) {
    {   // stage 32x32 bf16 A tile from hid (already bf16, rows contiguous)
      int i0 = tid * 4;
      int r  = i0 >> 5, k = i0 & 31;
      int rr = (r < nRows) ? r : 0;
      *(uint2*)&sA[r * 32 + k] =
          *(const uint2*)&hid[(size_t)(rowBase + rr) * H_MID + kb + k];
    }
    __syncthreads();

    BPair ap, ahp;
    ap.lo  = *(const uint4*)&sA[cn * 32 + h8 * 8];
    ap.hi  = *(const uint4*)&sA[cn * 32 + 16 + h8 * 8];
    ahp.lo = *(const uint4*)&sA[(16 + cn) * 32 + h8 * 8];
    ahp.hi = *(const uint4*)&sA[(16 + cn) * 32 + 16 + h8 * 8];
    v16bf aLo = __builtin_bit_cast(v16bf, ap);
    v16bf aHi = __builtin_bit_cast(v16bf, ahp);

    v16bf bF[4];
#pragma unroll
    for (int s = 0; s < 4; ++s) {
      const unsigned short* bp =
          W2t + ((size_t)(e * O_OUT + colBase + s * 16 + cn)) * H_MID + kb + h8 * 8;
      __builtin_prefetch((const void*)(bp + 32), 0, 0);
      BPair bb;
      bb.lo = *(const uint4*)bp;
      bb.hi = *(const uint4*)(bp + 16);
      bF[s] = __builtin_bit_cast(v16bf, bb);
    }
#pragma unroll
    for (int s = 0; s < 4; ++s) {
      accL[s] = __builtin_amdgcn_wmma_f32_16x16x32_bf16(
          false, aLo, false, bF[s], (short)0, accL[s], false, false);
      accH[s] = __builtin_amdgcn_wmma_f32_16x16x32_bf16(
          false, aHi, false, bF[s], (short)0, accH[s], false, false);
    }
    __syncthreads();
  }

#pragma unroll
  for (int s = 0; s < 4; ++s) {
    F8 frL = __builtin_bit_cast(F8, accL[s]);
    F8 frH = __builtin_bit_cast(F8, accH[s]);
    int ocol = colBase + s * 16 + cn;
#pragma unroll
    for (int i = 0; i < 8; ++i) {
      int rL = i + 8 * h8;
      int rH = 16 + rL;
      if (rL < nRows)
        out[(size_t)sTok[rL] * OUTW + ocol] = frL.f[i];
      if (rH < nRows)
        out[(size_t)sTok[rH] * OUTW + ocol] = frH.f[i];
    }
  }
}

// ---------------- one-hot tail columns ----------------
__global__ void k_onehot(const int* __restrict__ eidx, float* __restrict__ out) {
  int i = blockIdx.x * blockDim.x + threadIdx.x;
  if (i < N_TOK * E_EXP) {
    int n = i >> 3, j = i & 7;
    out[(size_t)n * OUTW + O_OUT + j] = (eidx[n] == j) ? 1.0f : 0.0f;
  }
}

// ---------------- launch ----------------
extern "C" void kernel_launch(void* const* d_in, const int* in_sizes, int n_in,
                              void* d_out, int out_size, void* d_ws, size_t ws_size,
                              hipStream_t stream) {
  (void)in_sizes; (void)n_in; (void)out_size; (void)ws_size;
  const float* x    = (const float*)d_in[0];
  const float* W1   = (const float*)d_in[1];
  const float* b1   = (const float*)d_in[2];
  const float* W2   = (const float*)d_in[3];
  const float* b2   = (const float*)d_in[4];
  const int*   eidx = (const int*)  d_in[5];
  float* out = (float*)d_out;

  char* w = (char*)d_ws;
  int* cnt  = (int*)(w + 0);
  int* off  = (int*)(w + 64);
  int* tOff = (int*)(w + 128);
  int* slot = (int*)(w + 256);
  int* perm = (int*)(w + 256 + 4 * N_TOK);
  const size_t W1T_OFF = 1u << 17;
  const size_t W1T_BYTES = (size_t)2 * E_EXP * D_IN * H_MID;   // 32 MiB
  const size_t W2T_BYTES = (size_t)2 * E_EXP * H_MID * O_OUT;  // 32 MiB
  unsigned short* W1t = (unsigned short*)(w + W1T_OFF);
  unsigned short* W2t = (unsigned short*)(w + W1T_OFF + W1T_BYTES);
  unsigned short* hid = (unsigned short*)(w + W1T_OFF + W1T_BYTES + W2T_BYTES);

  k_init_cnt<<<1, 32, 0, stream>>>(cnt);
  k_count  <<<N_TOK / 256, 256, 0, stream>>>(eidx, cnt, slot);
  k_scan   <<<1, 1, 0, stream>>>(cnt, off, tOff);
  k_scatter<<<N_TOK / 256, 256, 0, stream>>>(eidx, slot, off, perm);

  k_cvt_w1 <<<(E_EXP * D_IN * H_MID) / 256, 256, 0, stream>>>(W1, W1t);
  k_cvt_w2 <<<(E_EXP * H_MID * O_OUT) / 256, 256, 0, stream>>>(W2, W2t);

  dim3 g1(H_MID / 512, MAX_TILES);
  k_gemm1<<<g1, 256, 0, stream>>>(x, W1t, b1, perm, off, tOff, hid);
  dim3 g2(O_OUT / 512, MAX_TILES);
  k_gemm2<<<g2, 256, 0, stream>>>(hid, W2t, b2, perm, off, tOff, out);

  k_onehot<<<(N_TOK * E_EXP) / 256, 256, 0, stream>>>(eidx, out);
}
